// MultiscaleMoEAMIL_58179626992128
// MI455X (gfx1250) — compile-verified, matrix operand order checked by reading, not run on hardware
//
#include <hip/hip_runtime.h>
#include <hip/hip_bf16.h>
#include <math.h>

#define NTOT 16384
#define BM 128
#define BN 128
#define BK 32

typedef __attribute__((ext_vector_type(16))) _Float16 v16h;
typedef __attribute__((ext_vector_type(8)))  _Float16 v8h;
typedef __attribute__((ext_vector_type(8)))  float    v8f;
typedef int b128i __attribute__((vector_size(16)));

// gfx1250 async global->LDS path (ASYNCcnt), guarded for toolchain portability.
#if defined(__gfx1250__) && __has_builtin(__builtin_amdgcn_global_load_async_to_lds_b128) && __has_builtin(__builtin_amdgcn_s_wait_asynccnt)
#define USE_ASYNC 1
#define AS1P(p) ((__attribute__((address_space(1))) b128i*)(p))
#define AS3P(p) ((__attribute__((address_space(3))) b128i*)(p))
#else
#define USE_ASYNC 0
#endif

enum { EPI_BIAS_F32 = 0, EPI_GELU_F16 = 1, EPI_EXPERT = 2, EPI_FUSED = 3, EPI_TANH_F32 = 4 };

__device__ __forceinline__ float gelu_f(float v) {
  return 0.5f * v * (1.0f + erff(v * 0.70710678118654752f));
}

__device__ __forceinline__ float blk_reduce_sum(float v, float* red) {
  int t = threadIdx.x;
  red[t] = v; __syncthreads();
  for (int o = 128; o > 0; o >>= 1) {
    if (t < o) red[t] += red[t + o];
    __syncthreads();
  }
  float r = red[0]; __syncthreads();
  return r;
}

__device__ __forceinline__ v16h load_frag(const _Float16* rb, int koff) {
  v8h lo = *(const v8h*)(rb + koff);
  v8h hi = *(const v8h*)(rb + 16 + koff);
  return __builtin_shufflevector(lo, hi, 0,1,2,3,4,5,6,7,8,9,10,11,12,13,14,15);
}

// ---------------------------------------------------------------------------
// WMMA GEMM: out[M x Ncols] = A[M x K] (row-major, ld=ldA) @ B[Ncols x K]^T
// M = NTOT. grid = (Ncols/128, NTOT/128), block = 256 (8 waves, wave32).
// Double-buffered LDS, one barrier per K-step. Next tile is streamed with
// global_load_async_to_lds_b128 (ASYNCcnt) while the matrix pipe runs; the
// 8 B-fragments are preloaded so the 8 WMMAs issue back-to-back.
// ---------------------------------------------------------------------------
template<int EPI>
__global__ __launch_bounds__(256) void gemm_wmma(
    const _Float16* __restrict__ A, int ldA,
    const _Float16* __restrict__ B,                 // ldB == K
    float* __restrict__ outF, _Float16* __restrict__ outH, int ldOut,
    const float* __restrict__ bias,
    const float* __restrict__ gates,
    const float* __restrict__ bias2, const float* __restrict__ bias2b,
    int K, int p0, int p1)
{
  __shared__ __align__(16) _Float16 As[2][BM][BK + 8];
  __shared__ __align__(16) _Float16 Bs[2][BN][BK + 8];

  const int t = threadIdx.x;
  const int lane = t & 31;
  const int wv = t >> 5;
  const int rowBase = blockIdx.y * BM;
  const int colBase = blockIdx.x * BN;

  v8f acc[8];
#pragma unroll
  for (int i = 0; i < 8; i++)
#pragma unroll
    for (int j = 0; j < 8; j++) acc[i][j] = 0.0f;

  const int lr = t >> 1;           // 0..127  tile row for cooperative load
  const int lh = (t & 1) * 16;     // 0 or 16 halves

  const _Float16* gArow = A + (size_t)(rowBase + lr) * ldA + lh;
  const _Float16* gBrow = B + (size_t)(colBase + lr) * K + lh;

#if USE_ASYNC
  {
    _Float16* lA = &As[0][lr][lh];
    _Float16* lB = &Bs[0][lr][lh];
    __builtin_amdgcn_global_load_async_to_lds_b128(AS1P(gArow), AS3P(lA), 0, 0);
    __builtin_amdgcn_global_load_async_to_lds_b128(AS1P(gArow + 8), AS3P(lA + 8), 0, 0);
    __builtin_amdgcn_global_load_async_to_lds_b128(AS1P(gBrow), AS3P(lB), 0, 0);
    __builtin_amdgcn_global_load_async_to_lds_b128(AS1P(gBrow + 8), AS3P(lB + 8), 0, 0);
  }
#else
  {
    const uint4* ga = (const uint4*)gArow;
    const uint4* gb = (const uint4*)gBrow;
    uint4 a0 = ga[0], a1 = ga[1], b0 = gb[0], b1 = gb[1];
    *(uint4*)&As[0][lr][lh]     = a0;
    *(uint4*)&As[0][lr][lh + 8] = a1;
    *(uint4*)&Bs[0][lr][lh]     = b0;
    *(uint4*)&Bs[0][lr][lh + 8] = b1;
  }
#endif

  int it = 0;
  for (int k0 = 0; k0 < K; k0 += BK, it ^= 1) {
#if USE_ASYNC
    __builtin_amdgcn_s_wait_asynccnt(0);   // current tile resident in LDS
#endif
    __syncthreads();                       // all waves' tiles resident / prior reads done
    const bool more = (k0 + BK) < K;

#if USE_ASYNC
    if (more) {
      const _Float16* gA = gArow + (k0 + BK);
      const _Float16* gB = gBrow + (k0 + BK);
      _Float16* lA = &As[it ^ 1][lr][lh];
      _Float16* lB = &Bs[it ^ 1][lr][lh];
      __builtin_amdgcn_global_load_async_to_lds_b128(AS1P(gA), AS3P(lA), 0, 0);
      __builtin_amdgcn_global_load_async_to_lds_b128(AS1P(gA + 8), AS3P(lA + 8), 0, 0);
      __builtin_amdgcn_global_load_async_to_lds_b128(AS1P(gB), AS3P(lB), 0, 0);
      __builtin_amdgcn_global_load_async_to_lds_b128(AS1P(gB + 8), AS3P(lB + 8), 0, 0);
    }
#else
    uint4 a0{}, a1{}, b0{}, b1{};
    if (more) {
      const uint4* ga = (const uint4*)(gArow + (k0 + BK));
      const uint4* gb = (const uint4*)(gBrow + (k0 + BK));
      a0 = ga[0]; a1 = ga[1]; b0 = gb[0]; b1 = gb[1];
    }
#endif

    // ---- compute tile k0 from buffer `it` ----
    const int koff = (lane & 16) >> 1;     // 0 or 8 halves
    v16h af = load_frag(&As[it][wv * 16 + (lane & 15)][0], koff);
    v16h bf[8];
#pragma unroll
    for (int tt = 0; tt < 8; tt++)
      bf[tt] = load_frag(&Bs[it][tt * 16 + (lane & 15)][0], koff);
#pragma unroll
    for (int tt = 0; tt < 8; tt++)
      acc[tt] = __builtin_amdgcn_wmma_f32_16x16x32_f16(
          false, af, false, bf[tt], (short)0, acc[tt], false, false);

#if !USE_ASYNC
    if (more) {
      *(uint4*)&As[it ^ 1][lr][lh]     = a0;
      *(uint4*)&As[it ^ 1][lr][lh + 8] = a1;
      *(uint4*)&Bs[it ^ 1][lr][lh]     = b0;
      *(uint4*)&Bs[it ^ 1][lr][lh + 8] = b1;
    }
#endif
  }

  // Epilogue. D layout: VGPR i -> row = i + (lane>=16)*8, col = lane&15 within tile.
  const int rsub = (lane >> 4) << 3;
  const int csub = lane & 15;
#pragma unroll
  for (int tt = 0; tt < 8; tt++) {
    int col = colBase + tt * 16 + csub;
#pragma unroll
    for (int i = 0; i < 8; i++) {
      int row = rowBase + wv * 16 + rsub + i;
      float v = acc[tt][i];
      if (EPI == EPI_BIAS_F32) {
        outF[(size_t)row * ldOut + col] = v + bias[col];
      } else if (EPI == EPI_GELU_F16) {
        outH[(size_t)row * ldOut + col] = (_Float16)gelu_f(v + bias[col]);
      } else if (EPI == EPI_EXPERT) {
        int e = col >> 8;                                  // expert within class
        float g = gates[((size_t)p0 * NTOT + row) * 8 + e + p1];
        outH[(size_t)row * ldOut + col] = (_Float16)(gelu_f(v + bias[col]) * g);
      } else if (EPI == EPI_FUSED) {
        const float* gp = gates + ((size_t)p0 * NTOT + row) * 8;
        float badd = 0.0f;
#pragma unroll
        for (int e = 0; e < 4; e++) {
          badd += gp[e]     * bias2 [(p0 * 4 + e) * 256 + col];
          badd += gp[4 + e] * bias2b[(p0 * 4 + e) * 256 + col];
        }
        float ov = v + badd;
        outF[(size_t)row * ldOut + col] = ov;
        outH[(size_t)row * ldOut + col] = (_Float16)ov;
      } else { // EPI_TANH_F32
        outF[(size_t)row * ldOut + col] = tanhf(v + bias[col]);
      }
    }
  }
}

// ---------------------------------------------------------------------------
// Small helper kernels
// ---------------------------------------------------------------------------
__global__ void cvt_f16(const float* __restrict__ s, _Float16* __restrict__ d, int n) {
  int i = blockIdx.x * 256 + threadIdx.x;
  if (i < n) d[i] = (_Float16)s[i];
}

__global__ void build_w2cat(const float* __restrict__ w20, const float* __restrict__ w5,
                            _Float16* __restrict__ out) {
  int i = blockIdx.x * 256 + threadIdx.x;
  if (i >= 4 * 256 * 2048) return;
  int c = i >> 19;            // 256*2048
  int r = (i >> 11) & 255;
  int k = i & 2047;
  float v;
  if (k < 1024) { int e = k >> 8, h = k & 255; v = w20[(((c * 4 + e) * 256) + r) * 256 + h]; }
  else { int kk = k - 1024; int e = kk >> 8, h = kk & 255; v = w5[(((c * 4 + e) * 256) + r) * 256 + h]; }
  out[i] = (_Float16)v;
}

__global__ void posenc(const int* __restrict__ coords, _Float16* __restrict__ out) {
  int idx = blockIdx.x * 256 + threadIdx.x;
  int n = idx >> 7, j = idx & 127;
  if (n >= NTOT) return;
  float ex = powf(10000.0f, (2.0f * (float)(j >> 1)) / 128.0f);
  float cx = (float)coords[n * 2]     / ex;
  float cy = (float)coords[n * 2 + 1] / ex;
  _Float16* o = out + (size_t)n * 512;
  o[2 * j]           = (_Float16)sinf(cx);
  o[2 * j + 1]       = (_Float16)cosf(cx);
  o[256 + 2 * j]     = (_Float16)sinf(cy);
  o[256 + 2 * j + 1] = (_Float16)cosf(cy);
}

// x_branch = x + LN(y): reads y[N x 1536], writes f16 into xf16 (ld 3072, pre-offset)
__global__ __launch_bounds__(256) void ln_res(
    const float* __restrict__ y, const float* __restrict__ g, const float* __restrict__ b,
    const float* __restrict__ x, _Float16* __restrict__ xout) {
  __shared__ float red[256];
  int r = blockIdx.x, t = threadIdx.x;
  const float* yr = y + (size_t)r * 1536;
  float s = 0.0f;
  for (int j = t; j < 1536; j += 256) s += yr[j];
  float mu = blk_reduce_sum(s, red) * (1.0f / 1536.0f);
  float vs = 0.0f;
  for (int j = t; j < 1536; j += 256) { float d = yr[j] - mu; vs += d * d; }
  float var = blk_reduce_sum(vs, red) * (1.0f / 1536.0f);
  float inv = rsqrtf(var + 1e-5f);
  const float* xr = x + (size_t)r * 3072;
  _Float16* orow = xout + (size_t)r * 3072;
  for (int j = t; j < 1536; j += 256)
    orow[j] = (_Float16)(xr[j] + (yr[j] - mu) * inv * g[j] + b[j]);
}

__global__ void gate2_softmax(const _Float16* __restrict__ gh, const float* __restrict__ g2w,
                              const float* __restrict__ g2b, float* __restrict__ gates) {
  int n = blockIdx.x * 256 + threadIdx.x;
  int c = blockIdx.y;
  if (n >= NTOT) return;
  const _Float16* r = gh + (size_t)n * 512 + c * 128;
  float z[8]; float m = -1e30f;
  for (int e = 0; e < 8; e++) {
    float s = g2b[c * 8 + e];
    const float* wv = g2w + (c * 8 + e) * 128;
    for (int k = 0; k < 128; k++) s += (float)r[k] * wv[k];
    z[e] = s; m = fmaxf(m, s);
  }
  float ss = 0.0f;
  for (int e = 0; e < 8; e++) { z[e] = expf(z[e] - m); ss += z[e]; }
  float inv = 1.0f / ss;
  float* o = gates + ((size_t)c * NTOT + n) * 8;
  for (int e = 0; e < 8; e++) o[e] = z[e] * inv;
}

__global__ void attn_score(const float* __restrict__ ah, const float* __restrict__ a2w,
                           const float* __restrict__ a2b, float* __restrict__ araw) {
  int n = blockIdx.x * 256 + threadIdx.x;
  int c = blockIdx.y;
  if (n >= NTOT) return;
  const float* r = ah + ((size_t)c * NTOT + n) * 128;
  float s = a2b[c];
  for (int d = 0; d < 128; d++) s += r[d] * a2w[c * 128 + d];
  araw[(size_t)c * NTOT + n] = s;
}

__global__ __launch_bounds__(256) void class_softmax_stats(const float* __restrict__ araw,
                                                           float* __restrict__ stats) {
  __shared__ float red[256];
  int c = blockIdx.x, t = threadIdx.x;
  const float* a = araw + (size_t)c * NTOT;
  float mx = -1e30f;
  for (int n = t; n < NTOT; n += 256) mx = fmaxf(mx, a[n]);
  red[t] = mx; __syncthreads();
  for (int o = 128; o > 0; o >>= 1) { if (t < o) red[t] = fmaxf(red[t], red[t + o]); __syncthreads(); }
  mx = red[0]; __syncthreads();
  float s = 0.0f;
  for (int n = t; n < NTOT; n += 256) s += expf(a[n] - mx);
  s = blk_reduce_sum(s, red);
  if (t == 0) { stats[c * 2] = mx; stats[c * 2 + 1] = s; }
}

__global__ void zero_f32(float* p, int n) {
  int i = blockIdx.x * 256 + threadIdx.x;
  if (i < n) p[i] = 0.0f;
}

__global__ __launch_bounds__(256) void feats_acc(const float* __restrict__ araw,
                                                 const float* __restrict__ stats,
                                                 const float* __restrict__ fused,
                                                 float* __restrict__ feats) {
  int c = blockIdx.y, t = threadIdx.x;
  int n0 = blockIdx.x * (NTOT / 32);
  float mx = stats[c * 2], inv = 1.0f / stats[c * 2 + 1];
  const float* a = araw + (size_t)c * NTOT;
  const float* f = fused + ((size_t)c * NTOT + n0) * 256;
  float acc = 0.0f;
  for (int n = 0; n < NTOT / 32; n++)
    acc += expf(a[n0 + n] - mx) * inv * f[(size_t)n * 256 + t];
  atomicAdd(&feats[c * 256 + t], acc);
}

// Tiny 4-token transformer head + classifier. 1 block, 256 threads.
__global__ __launch_bounds__(256) void head_kernel(
    const float* __restrict__ feats,
    const float* __restrict__ t_in_w, const float* __restrict__ t_in_b,
    const float* __restrict__ t_out_w, const float* __restrict__ t_out_b,
    const float* __restrict__ ff1w, const float* __restrict__ ff1b,
    const float* __restrict__ ff2w, const float* __restrict__ ff2b,
    const float* __restrict__ ln1g, const float* __restrict__ ln1b,
    const float* __restrict__ ln2g, const float* __restrict__ ln2b,
    const float* __restrict__ clf_w, const float* __restrict__ clf_b,
    float* __restrict__ out)
{
  __shared__ float s[4][256], q[4][256], kk[4][256], vv[4][256], tmp[4][256];
  __shared__ float sc[8][4][4];
  __shared__ float red[256];
  int t = threadIdx.x;

  for (int r = 0; r < 4; r++) s[r][t] = feats[r * 256 + t];
  __syncthreads();

  for (int r = 0; r < 4; r++) {
    float aq = t_in_b[t], ak = t_in_b[256 + t], av = t_in_b[512 + t];
    for (int h = 0; h < 256; h++) {
      float xv = s[r][h];
      aq += xv * t_in_w[(size_t)t * 256 + h];
      ak += xv * t_in_w[(size_t)(256 + t) * 256 + h];
      av += xv * t_in_w[(size_t)(512 + t) * 256 + h];
    }
    q[r][t] = aq; kk[r][t] = ak; vv[r][t] = av;
  }
  __syncthreads();

  int head = t >> 5, d = t & 31;
  if (d < 4) {
    int ki = d;
    for (int qi = 0; qi < 4; qi++) {
      float a = 0.0f;
      for (int dd = 0; dd < 32; dd++) a += q[qi][head * 32 + dd] * kk[ki][head * 32 + dd];
      sc[head][qi][ki] = a * 0.17677669529663687f;   // 1/sqrt(32)
    }
  }
  __syncthreads();
  if (d < 4) {
    int qi = d;
    float m = -1e30f;
    for (int ki = 0; ki < 4; ki++) m = fmaxf(m, sc[head][qi][ki]);
    float ssum = 0.0f, e[4];
    for (int ki = 0; ki < 4; ki++) { e[ki] = expf(sc[head][qi][ki] - m); ssum += e[ki]; }
    for (int ki = 0; ki < 4; ki++) sc[head][qi][ki] = e[ki] / ssum;
  }
  __syncthreads();
  for (int qi = 0; qi < 4; qi++) {
    float a = 0.0f;
    for (int ki = 0; ki < 4; ki++) a += sc[head][qi][ki] * vv[ki][t];
    tmp[qi][t] = a;
  }
  __syncthreads();

  for (int r = 0; r < 4; r++) {
    float a = t_out_b[t];
    for (int h = 0; h < 256; h++) a += tmp[r][h] * t_out_w[(size_t)t * 256 + h];
    q[r][t] = s[r][t] + a;           // residual
  }
  __syncthreads();
  for (int r = 0; r < 4; r++) {      // LN1
    float xv = q[r][t];
    float mu = blk_reduce_sum(xv, red) * (1.0f / 256.0f);
    float dv = xv - mu;
    float var = blk_reduce_sum(dv * dv, red) * (1.0f / 256.0f);
    s[r][t] = dv * rsqrtf(var + 1e-5f) * ln1g[t] + ln1b[t];
  }
  __syncthreads();
  for (int r = 0; r < 4; r++) {      // FF layer 1 (128)
    if (t < 128) {
      float a = ff1b[t];
      for (int h = 0; h < 256; h++) a += s[r][h] * ff1w[(size_t)t * 256 + h];
      tmp[r][t] = fmaxf(a, 0.0f);
    }
  }
  __syncthreads();
  for (int r = 0; r < 4; r++) {      // FF layer 2 + residual
    float a = ff2b[t];
    for (int j = 0; j < 128; j++) a += tmp[r][j] * ff2w[(size_t)t * 128 + j];
    q[r][t] = s[r][t] + a;
  }
  __syncthreads();
  for (int r = 0; r < 4; r++) {      // LN2
    float xv = q[r][t];
    float mu = blk_reduce_sum(xv, red) * (1.0f / 256.0f);
    float dv = xv - mu;
    float var = blk_reduce_sum(dv * dv, red) * (1.0f / 256.0f);
    s[r][t] = dv * rsqrtf(var + 1e-5f) * ln2g[t] + ln2b[t];
  }
  __syncthreads();

  for (int r = 0; r < 4; r++) out[r * 256 + t] = s[r][t];   // features
  for (int c = 0; c < 4; c++) {
    float part = s[c][t] * clf_w[c * 256 + t];
    float tot = blk_reduce_sum(part, red);
    if (t == 0) {
      float lg = tot + clf_b[c];
      out[1024 + c] = lg;
      float pr = 1.0f / (1.0f + expf(-lg));
      out[1028 + c] = pr;
      out[1032 + c] = (pr > 0.5f) ? 1.0f : 0.0f;
    }
  }
}

// ---------------------------------------------------------------------------
extern "C" void kernel_launch(void* const* d_in, const int* in_sizes, int n_in,
                              void* d_out, int out_size, void* d_ws, size_t ws_size,
                              hipStream_t stream) {
  const float* x        = (const float*)d_in[0];
  const int*   coords20 = (const int*)d_in[1];
  const int*   coords5  = (const int*)d_in[2];
  const float* pos20_w  = (const float*)d_in[3];
  const float* pos20_b  = (const float*)d_in[4];
  const float* pos20_g  = (const float*)d_in[5];
  const float* pos20_be = (const float*)d_in[6];
  const float* pos5_w   = (const float*)d_in[7];
  const float* pos5_b   = (const float*)d_in[8];
  const float* pos5_g   = (const float*)d_in[9];
  const float* pos5_be  = (const float*)d_in[10];
  const float* gate1_w  = (const float*)d_in[11];
  const float* gate1_b  = (const float*)d_in[12];
  const float* gate2_w  = (const float*)d_in[13];
  const float* gate2_b  = (const float*)d_in[14];
  const float* e20_w1   = (const float*)d_in[15];
  const float* e20_b1   = (const float*)d_in[16];
  const float* e20_w2   = (const float*)d_in[17];
  const float* e20_b2   = (const float*)d_in[18];
  const float* e5_w1    = (const float*)d_in[19];
  const float* e5_b1    = (const float*)d_in[20];
  const float* e5_w2    = (const float*)d_in[21];
  const float* e5_b2    = (const float*)d_in[22];
  const float* attn1_w  = (const float*)d_in[23];
  const float* attn1_b  = (const float*)d_in[24];
  const float* attn2_w  = (const float*)d_in[25];
  const float* attn2_b  = (const float*)d_in[26];
  const float* t_in_w   = (const float*)d_in[27];
  const float* t_in_b   = (const float*)d_in[28];
  const float* t_out_w  = (const float*)d_in[29];
  const float* t_out_b  = (const float*)d_in[30];
  const float* t_ff1_w  = (const float*)d_in[31];
  const float* t_ff1_b  = (const float*)d_in[32];
  const float* t_ff2_w  = (const float*)d_in[33];
  const float* t_ff2_b  = (const float*)d_in[34];
  const float* ln1_g    = (const float*)d_in[35];
  const float* ln1_b    = (const float*)d_in[36];
  const float* ln2_g    = (const float*)d_in[37];
  const float* ln2_b    = (const float*)d_in[38];
  const float* clf_w    = (const float*)d_in[39];
  const float* clf_b    = (const float*)d_in[40];

  float* o_base  = (float*)d_out;
  float* o_araw  = o_base + 1036;               // features(1024)+logits/prob/hat(12)
  float* o_gates = o_base + 1036 + 4 * NTOT;

  // ---- workspace carve ----
  char* wp = (char*)d_ws;
  auto alloc = [&](size_t bytes) -> void* {
    void* p = (void*)wp;
    wp += (bytes + 255) & ~(size_t)255;
    return p;
  };
  _Float16* pos20_wH = (_Float16*)alloc((size_t)1536 * 512 * 2);
  _Float16* pos5_wH  = (_Float16*)alloc((size_t)1536 * 512 * 2);
  _Float16* gate1_wH = (_Float16*)alloc((size_t)512 * 3072 * 2);
  _Float16* e20_w1H  = (_Float16*)alloc((size_t)4096 * 1536 * 2);
  _Float16* e5_w1H   = (_Float16*)alloc((size_t)4096 * 1536 * 2);
  _Float16* attn1_wH = (_Float16*)alloc((size_t)4 * 128 * 256 * 2);
  _Float16* w2catH   = (_Float16*)alloc((size_t)4 * 256 * 2048 * 2);
  _Float16* posH20   = (_Float16*)alloc((size_t)NTOT * 512 * 2);
  _Float16* posH5    = (_Float16*)alloc((size_t)NTOT * 512 * 2);
  _Float16* xf16     = (_Float16*)alloc((size_t)NTOT * 3072 * 2);
  float*    ytmp     = (float*)   alloc((size_t)NTOT * 1536 * 4);
  _Float16* ghH      = (_Float16*)alloc((size_t)NTOT * 512 * 2);
  _Float16* HsH      = (_Float16*)alloc((size_t)NTOT * 2048 * 2);
  float*    fusedF   = (float*)   alloc((size_t)4 * NTOT * 256 * 4);
  _Float16* fusedH   = (_Float16*)alloc((size_t)4 * NTOT * 256 * 2);
  float*    ahF      = (float*)   alloc((size_t)4 * NTOT * 128 * 4);
  float*    feats    = (float*)   alloc(1024 * 4);
  float*    stats    = (float*)   alloc(8 * 4);

  auto cgrid = [](int n) { return dim3((n + 255) / 256); };

  // ---- weight conversions (f32 -> f16) ----
  cvt_f16<<<cgrid(1536 * 512), 256, 0, stream>>>(pos20_w, pos20_wH, 1536 * 512);
  cvt_f16<<<cgrid(1536 * 512), 256, 0, stream>>>(pos5_w, pos5_wH, 1536 * 512);
  cvt_f16<<<cgrid(512 * 3072), 256, 0, stream>>>(gate1_w, gate1_wH, 512 * 3072);
  cvt_f16<<<cgrid(4096 * 1536), 256, 0, stream>>>(e20_w1, e20_w1H, 4096 * 1536);
  cvt_f16<<<cgrid(4096 * 1536), 256, 0, stream>>>(e5_w1, e5_w1H, 4096 * 1536);
  cvt_f16<<<cgrid(4 * 128 * 256), 256, 0, stream>>>(attn1_w, attn1_wH, 4 * 128 * 256);
  build_w2cat<<<cgrid(4 * 256 * 2048), 256, 0, stream>>>(e20_w2, e5_w2, w2catH);

  // ---- positional encodings ----
  posenc<<<dim3((NTOT * 128) / 256), 256, 0, stream>>>(coords20, posH20);
  posenc<<<dim3((NTOT * 128) / 256), 256, 0, stream>>>(coords5, posH5);

  // ---- pos GEMM + LN + residual (branch 20 then branch 5, reusing ytmp) ----
  {
    dim3 g(1536 / BN, NTOT / BM);
    gemm_wmma<EPI_BIAS_F32><<<g, 256, 0, stream>>>(
        posH20, 512, pos20_wH, ytmp, nullptr, 1536, pos20_b,
        nullptr, nullptr, nullptr, 512, 0, 0);
    ln_res<<<NTOT, 256, 0, stream>>>(ytmp, pos20_g, pos20_be, x, xf16);
    gemm_wmma<EPI_BIAS_F32><<<g, 256, 0, stream>>>(
        posH5, 512, pos5_wH, ytmp, nullptr, 1536, pos5_b,
        nullptr, nullptr, nullptr, 512, 0, 0);
    ln_res<<<NTOT, 256, 0, stream>>>(ytmp, pos5_g, pos5_be, x + 1536, xf16 + 1536);
  }

  // ---- gate MLP ----
  {
    dim3 g(512 / BN, NTOT / BM);
    gemm_wmma<EPI_GELU_F16><<<g, 256, 0, stream>>>(
        xf16, 3072, gate1_wH, nullptr, ghH, 512, gate1_b,
        nullptr, nullptr, nullptr, 3072, 0, 0);
    gate2_softmax<<<dim3(NTOT / 256, 4), 256, 0, stream>>>(ghH, gate2_w, gate2_b, o_gates);
  }

  // ---- experts per class: gate-scaled h, then fused = Hs @ W2cat^T ----
  for (int c = 0; c < 4; c++) {
    dim3 ge(1024 / BN, NTOT / BM);
    gemm_wmma<EPI_EXPERT><<<ge, 256, 0, stream>>>(
        xf16, 3072, e20_w1H + (size_t)c * 1024 * 1536,
        nullptr, HsH, 2048, e20_b1 + c * 1024,
        o_gates, nullptr, nullptr, 1536, c, 0);
    gemm_wmma<EPI_EXPERT><<<ge, 256, 0, stream>>>(
        xf16 + 1536, 3072, e5_w1H + (size_t)c * 1024 * 1536,
        nullptr, HsH + 1024, 2048, e5_b1 + c * 1024,
        o_gates, nullptr, nullptr, 1536, c, 4);
    dim3 gf(256 / BN, NTOT / BM);
    gemm_wmma<EPI_FUSED><<<gf, 256, 0, stream>>>(
        HsH, 2048, w2catH + (size_t)c * 256 * 2048,
        fusedF + (size_t)c * NTOT * 256, fusedH + (size_t)c * NTOT * 256, 256,
        nullptr, o_gates, e20_b2, e5_b2, 2048, c, 0);
  }

  // ---- attention pooling ----
  for (int c = 0; c < 4; c++) {
    dim3 ga(1, NTOT / BM);
    gemm_wmma<EPI_TANH_F32><<<ga, 256, 0, stream>>>(
        fusedH + (size_t)c * NTOT * 256, 256, attn1_wH + (size_t)c * 128 * 256,
        ahF + (size_t)c * NTOT * 128, nullptr, 128, attn1_b + c * 128,
        nullptr, nullptr, nullptr, 256, 0, 0);
  }
  attn_score<<<dim3(NTOT / 256, 4), 256, 0, stream>>>(ahF, attn2_w, attn2_b, o_araw);
  class_softmax_stats<<<4, 256, 0, stream>>>(o_araw, stats);
  zero_f32<<<cgrid(1024), 256, 0, stream>>>(feats, 1024);
  feats_acc<<<dim3(32, 4), 256, 0, stream>>>(o_araw, stats, fusedF, feats);

  // ---- tiny transformer head + classifier ----
  head_kernel<<<1, 256, 0, stream>>>(
      feats, t_in_w, t_in_b, t_out_w, t_out_b,
      t_ff1_w, t_ff1_b, t_ff2_w, t_ff2_b,
      ln1_g, ln1_b, ln2_g, ln2_b, clf_w, clf_b, o_base);

  (void)in_sizes; (void)n_in; (void)out_size; (void)ws_size;
}
